// ParallelAttention_16578573763115
// MI455X (gfx1250) — compile-verified, matrix operand order checked by reading, not run on hardware
//
#include <hip/hip_runtime.h>
#include <hip/hip_bf16.h>
#include <math.h>

typedef __attribute__((ext_vector_type(16))) _Float16 v16h;
typedef __attribute__((ext_vector_type(8)))  _Float16 v8h;
typedef __attribute__((ext_vector_type(8)))  float    v8f;
typedef __attribute__((ext_vector_type(4)))  int      v4i;

// Problem dimensions
constexpr int CB  = 256;   // batch
constexpr int CTQ = 64;    // question tokens
constexpr int CN  = 196;   // image positions
constexpr int CNP = 224;   // N padded to multiple of 32
constexpr int CE  = 1024;  // embed dim
constexpr int CK  = 512;   // attention dim

// Epilogue modes
constexpr int EP_BIAS_F16      = 0;  // D + bias -> f16 row-major
constexpr int EP_BIAS_F32_F16T = 1;  // D + bias -> f32 row-major and f16 transposed
constexpr int EP_TANH_WEIGHT   = 2;  // tanh(D), zero cols >= Nvalid -> f16 + f16 transposed
constexpr int EP_TANH_DOT      = 3;  // tanh(D + C32) * wvec, row-reduce, atomicAdd score

#if __has_builtin(__builtin_amdgcn_global_load_async_to_lds_b128)
#define USE_ASYNC_LDS 1
#else
#define USE_ASYNC_LDS 0
#endif

#if USE_ASYNC_LDS
__device__ __forceinline__ void async_copy_b128(const _Float16* gsrc, _Float16* ldst) {
  __builtin_amdgcn_global_load_async_to_lds_b128(
      (__attribute__((address_space(1))) v4i*)gsrc,
      (__attribute__((address_space(3))) v4i*)ldst, 0, 0);
}
#endif

template <int N>
__device__ __forceinline__ void async_wait() {
#if __has_builtin(__builtin_amdgcn_s_wait_asynccnt)
  __builtin_amdgcn_s_wait_asynccnt(N);
#else
  if (N == 0)
    asm volatile("s_wait_asynccnt 0" ::: "memory");
  else
    asm volatile("s_wait_asynccnt 3" ::: "memory");
#endif
}

__device__ __forceinline__ v16h combine8(v8h lo, v8h hi) {
  return __builtin_shufflevector(lo, hi, 0, 1, 2, 3, 4, 5, 6, 7,
                                 8, 9, 10, 11, 12, 13, 14, 15);
}

// ---------------------------------------------------------------------------
// Elementwise f32 -> f16
__global__ __launch_bounds__(256) void cvt_f32_f16(const float* __restrict__ src,
                                                   _Float16* __restrict__ dst, long n) {
  long i = (long)blockIdx.x * blockDim.x + threadIdx.x;
  if (i < n) dst[i] = (_Float16)src[i];
}

// dst[c*R + r] = src[r*C + c]  (transpose-convert), R rows, C cols in src
__global__ __launch_bounds__(256) void cvt_transpose_f16(const float* __restrict__ src,
                                                         _Float16* __restrict__ dst,
                                                         int R, int C) {
  long i = (long)blockIdx.x * blockDim.x + threadIdx.x;
  if (i >= (long)R * C) return;
  int c = (int)(i % C);
  int r = (int)(i / C);
  dst[(long)c * R + r] = (_Float16)src[i];
}

// img_feat [B,E,N] -> imgT f16 [B,NP,E], zero padded rows
__global__ __launch_bounds__(256) void build_imgT(const float* __restrict__ img,
                                                  _Float16* __restrict__ dst) {
  long i = (long)blockIdx.x * blockDim.x + threadIdx.x;
  long total = (long)CB * CNP * CE;
  if (i >= total) return;
  int e = (int)(i % CE);
  long t = i / CE;
  int n = (int)(t % CNP);
  int b = (int)(t / CNP);
  float v = (n < CN) ? img[((long)b * CE + e) * CN + n] : 0.0f;
  dst[i] = (_Float16)v;
}

__global__ __launch_bounds__(256) void zero_f32(float* __restrict__ p, long n) {
  long i = (long)blockIdx.x * blockDim.x + threadIdx.x;
  if (i < n) p[i] = 0.0f;
}

// ---------------------------------------------------------------------------
// Batched WMMA GEMM: D = A(M x Kdim) * B(Kdim x Nout) per batch,
// where B is supplied TRANSPOSED in memory: BT is [Nout x Kdim] row-major.
// Block tile 64x128, 256 threads = 8 waves; each wave owns a 32x32 output
// region = 2x2 WMMA tiles. Double-buffered async global->LDS pipeline with
// the last K-tile peeled so the steady-state loop is branch-free:
//   stage tile k+1 (3 async copies) -> s_wait_asynccnt 3 -> barrier ->
//   8x ds_load_b128 + 4x v_wmma -> barrier.
// Out-of-range staging rows are CLAMPED (not guarded): garbage only reaches
// output rows/cols that the guarded epilogue never stores.
template <int EPI>
__global__ __launch_bounds__(256)
void gemm_wmma(const _Float16* __restrict__ A, int lda, long strideA,
               const _Float16* __restrict__ BT, int ldb, long strideB,
               int M, int Nout, int Kdim,
               const float* __restrict__ bias,
               _Float16* __restrict__ C16, int ldc16, long strideC16,
               float* __restrict__ C32, int ldc32, long strideC32,
               _Float16* __restrict__ CT16, int ldct, long strideCT,
               int Nvalid,
               const float* __restrict__ wvec,
               float* __restrict__ score, int ldscore) {
  __shared__ _Float16 As[2][64][32];
  __shared__ _Float16 Bs[2][128][32];

  const int b     = blockIdx.z;
  const int m_blk = blockIdx.x * 64;
  const int n_blk = blockIdx.y * 128;

  const int tid = threadIdx.x;
  const int w   = tid >> 5;
  const int l   = tid & 31;
  const int wm  = w & 1;    // 0..1 : 32-row group
  const int wn  = w >> 1;   // 0..3 : 32-col group
  const int lm  = l & 15;
  const int lh  = l >> 4;
  const int kb  = lh * 8;

  // Staging coordinates: thread covers one A row-chunk + two BT row-chunks,
  // 8 halves (16B) each. Rows clamped into range -> uniform async issue count.
  const int srow = tid >> 2;        // 0..63
  const int scol = (tid & 3) * 8;   // 0,8,16,24 (halves)
  int gm  = m_blk + srow;       if (gm  >= M)    gm  = M - 1;
  int gn0 = n_blk + srow;       if (gn0 >= Nout) gn0 = Nout - 1;
  int gn1 = n_blk + srow + 64;  if (gn1 >= Nout) gn1 = Nout - 1;
  const _Float16* aSrc  = A  + (long)b * strideA + (long)gm  * lda + scol;
  const _Float16* b0Src = BT + (long)b * strideB + (long)gn0 * ldb + scol;
  const _Float16* b1Src = BT + (long)b * strideB + (long)gn1 * ldb + scol;

  v8f acc[2][2] = {{{}, {}}, {{}, {}}};

  // One WMMA tile-step: 8 contiguous ds_load_b128 -> 4 WMMA
  auto compute_tile = [&](int c) {
    v16h af[2], bf[2];
#pragma unroll
    for (int i = 0; i < 2; ++i) {
      const int row = wm * 32 + i * 16 + lm;
      af[i] = combine8(*(const v8h*)&As[c][row][kb], *(const v8h*)&As[c][row][kb + 16]);
    }
#pragma unroll
    for (int j = 0; j < 2; ++j) {
      const int row = wn * 32 + j * 16 + lm;
      bf[j] = combine8(*(const v8h*)&Bs[c][row][kb], *(const v8h*)&Bs[c][row][kb + 16]);
    }
#pragma unroll
    for (int i = 0; i < 2; ++i)
#pragma unroll
      for (int j = 0; j < 2; ++j)
        acc[i][j] = __builtin_amdgcn_wmma_f32_16x16x32_f16(false, af[i], false, bf[j],
                                                           (short)0, acc[i][j], false, false);
  };

#if USE_ASYNC_LDS
  // ---- prologue: stage tile 0 into buffer 0 ----
  async_copy_b128(aSrc,  &As[0][srow][scol]);
  async_copy_b128(b0Src, &Bs[0][srow][scol]);
  async_copy_b128(b1Src, &Bs[0][srow + 64][scol]);

  // ---- steady state: all but the last K-tile (branch-free body) ----
  int cur = 0;
  for (int kt = 0; kt + 32 < Kdim; kt += 32) {
    const int nk = kt + 32;
    async_copy_b128(aSrc + nk,  &As[cur ^ 1][srow][scol]);
    async_copy_b128(b0Src + nk, &Bs[cur ^ 1][srow][scol]);
    async_copy_b128(b1Src + nk, &Bs[cur ^ 1][srow + 64][scol]);
    async_wait<3>();   // 3 oldest (current tile) complete; next 3 in flight
    __syncthreads();
    compute_tile(cur);
    __syncthreads();   // buffer 'cur' free for restaging next iteration
    cur ^= 1;
  }
  // ---- peeled last tile ----
  async_wait<0>();
  __syncthreads();
  compute_tile(cur);
#else
  for (int kt = 0; kt < Kdim; kt += 32) {
    *(v8h*)&As[0][srow][scol]      = *(const v8h*)(aSrc + kt);
    *(v8h*)&Bs[0][srow][scol]      = *(const v8h*)(b0Src + kt);
    *(v8h*)&Bs[0][srow + 64][scol] = *(const v8h*)(b1Src + kt);
    __syncthreads();
    compute_tile(0);
    __syncthreads();
  }
#endif

  // ---- epilogue: 2x2 tiles; lane holds rows gmbase..gmbase+7, col gn ----
#pragma unroll
  for (int i = 0; i < 2; ++i) {
#pragma unroll
    for (int j = 0; j < 2; ++j) {
      const int gmbase = m_blk + wm * 32 + i * 16 + lh * 8;  // multiple of 8
      const int gn     = n_blk + wn * 32 + j * 16 + lm;
      const v8f a = acc[i][j];

      if (EPI == EP_BIAS_F16) {
        if (gn < Nout && gmbase < M) {
          const float bv = bias[gn];
#pragma unroll
          for (int r = 0; r < 8; ++r)
            C16[(long)b * strideC16 + (long)(gmbase + r) * ldc16 + gn] = (_Float16)(a[r] + bv);
        }
      } else if (EPI == EP_BIAS_F32_F16T) {
        if (gn < Nout && gmbase < M) {
          const float bv = bias[gn];
          v8h tv;
#pragma unroll
          for (int r = 0; r < 8; ++r) {
            const float v = a[r] + bv;
            C32[(long)b * strideC32 + (long)(gmbase + r) * ldc32 + gn] = v;
            tv[r] = (_Float16)v;
          }
          // transposed f16: row gn, contiguous cols gmbase..gmbase+7
          *(v8h*)(CT16 + (long)b * strideCT + (long)gn * ldct + gmbase) = tv;
        }
      } else if (EPI == EP_TANH_WEIGHT) {
        if (gn < Nout && gmbase < M) {
          v8h tv;
#pragma unroll
          for (int r = 0; r < 8; ++r) {
            const float v = (gn < Nvalid) ? tanhf(a[r]) : 0.0f;
            C16[(long)b * strideC16 + (long)(gmbase + r) * ldc16 + gn] = (_Float16)v;
            tv[r] = (_Float16)v;
          }
          *(v8h*)(CT16 + (long)b * strideCT + (long)gn * ldct + gmbase) = tv;
        }
      } else {  // EP_TANH_DOT
        const bool ok = (gn < Nout) && (gmbase < M);
#pragma unroll
        for (int r = 0; r < 8; ++r) {
          float v = 0.0f;
          if (ok)
            v = tanhf(a[r] + C32[(long)b * strideC32 + (long)(gmbase + r) * ldc32 + gn]) * wvec[gn];
          // lanes 0..15 / 16..31 share the same output row -> reduce, one atomic
          v += __shfl_xor(v, 1, 16);
          v += __shfl_xor(v, 2, 16);
          v += __shfl_xor(v, 4, 16);
          v += __shfl_xor(v, 8, 16);
          if (lm == 0 && gmbase < M) atomicAdd(&score[(long)b * ldscore + gmbase + r], v);
        }
      }
    }
  }
}

// ---------------------------------------------------------------------------
// Per-batch: softmax over masked question scores, attended question feature
__global__ __launch_bounds__(256)
void ques_finalize(const float* __restrict__ score, const float* __restrict__ bqa_p,
                   const float* __restrict__ mask, const float* __restrict__ qf,
                   float* __restrict__ out) {
  __shared__ float att[CTQ];
  const int b = blockIdx.x;
  const int t = threadIdx.x;
  const float bqa = bqa_p[0];
  if (t < CTQ) att[t] = (score[(long)b * CTQ + t] + bqa) * mask[(long)b * CTQ + t];
  __syncthreads();
  if (t == 0) {
    float mx = -1e30f;
    for (int q = 0; q < CTQ; ++q) mx = fmaxf(mx, att[q]);
    float s = 0.0f;
    for (int q = 0; q < CTQ; ++q) { att[q] = expf(att[q] - mx); s += att[q]; }
    const float inv = 1.0f / s;
    for (int q = 0; q < CTQ; ++q) att[q] *= inv;
  }
  __syncthreads();
  for (int e = t; e < CE; e += 256) {
    float a = 0.0f;
    for (int q = 0; q < CTQ; ++q) a += att[q] * qf[((long)b * CTQ + q) * CE + e];
    out[(long)b * CE + e] = a;
  }
}

// Per-batch: softmax over image scores, attended image feature (img is [B,E,N])
__global__ __launch_bounds__(256)
void img_finalize(const float* __restrict__ score, const float* __restrict__ bia_p,
                  const float* __restrict__ img, float* __restrict__ out) {
  __shared__ float att[CN];
  const int b = blockIdx.x;
  const int t = threadIdx.x;
  const float bia = bia_p[0];
  for (int n = t; n < CN; n += 256) att[n] = score[(long)b * CNP + n] + bia;
  __syncthreads();
  if (t == 0) {
    float mx = -1e30f;
    for (int n = 0; n < CN; ++n) mx = fmaxf(mx, att[n]);
    float s = 0.0f;
    for (int n = 0; n < CN; ++n) { att[n] = expf(att[n] - mx); s += att[n]; }
    const float inv = 1.0f / s;
    for (int n = 0; n < CN; ++n) att[n] *= inv;
  }
  __syncthreads();
  for (int e = t; e < CE; e += 256) {
    float a = 0.0f;
    for (int n = 0; n < CN; ++n) a += att[n] * img[((long)b * CE + e) * CN + n];
    out[(long)CB * CE + (long)b * CE + e] = a;
  }
}

// ---------------------------------------------------------------------------
extern "C" void kernel_launch(void* const* d_in, const int* in_sizes, int n_in,
                              void* d_out, int out_size, void* d_ws, size_t ws_size,
                              hipStream_t stream) {
  (void)in_sizes; (void)out_size; (void)ws_size;
  if (n_in < 13) return;
  const float* ques_feat = (const float*)d_in[0];   // [B,TQ,E]
  const float* img_feat  = (const float*)d_in[1];   // [B,E,N]
  const float* mask      = (const float*)d_in[2];   // [B,TQ]
  const float* Wc        = (const float*)d_in[3];   // [E,E]
  const float* bc        = (const float*)d_in[4];   // [E]
  const float* Wq        = (const float*)d_in[5];   // [E,K]
  const float* bq        = (const float*)d_in[6];   // [K]
  const float* Wi        = (const float*)d_in[7];   // [E,K]
  const float* bi        = (const float*)d_in[8];   // [K]
  const float* wqa       = (const float*)d_in[9];   // [K]
  const float* bqa       = (const float*)d_in[10];  // scalar
  const float* wia       = (const float*)d_in[11];  // [K]
  const float* bia       = (const float*)d_in[12];  // scalar
  float* out = (float*)d_out;

  // Workspace carve-out
  char* ws = (char*)d_ws;
  size_t off = 0;
  auto alloc = [&](size_t bytes) -> void* {
    void* p = ws + off;
    off = (off + bytes + 255) & ~(size_t)255;
    return p;
  };
  _Float16* qf16   = (_Float16*)alloc((size_t)CB * CTQ * CE * 2);  // [B,TQ,E]
  _Float16* imgT16 = (_Float16*)alloc((size_t)CB * CNP * CE * 2);  // [B,NP,E]
  _Float16* WcT16  = (_Float16*)alloc((size_t)CE * CE * 2);        // [E,E]   (Wc^T)
  _Float16* WqT16  = (_Float16*)alloc((size_t)CK * CE * 2);        // [K,E]   (Wq^T)
  _Float16* WiT16  = (_Float16*)alloc((size_t)CK * CE * 2);        // [K,E]   (Wi^T)
  _Float16* corr16 = (_Float16*)alloc((size_t)CB * CNP * CE * 2);  // [B,NP,E]
  float*    qe32   = (float*)   alloc((size_t)CB * CTQ * CK * 4);  // [B,TQ,K]
  _Float16* qeT16  = (_Float16*)alloc((size_t)CB * CK * CTQ * 2);  // [B,K,TQ]
  float*    ie32   = (float*)   alloc((size_t)CB * CNP * CK * 4);  // [B,NP,K]
  _Float16* ieT16  = (_Float16*)alloc((size_t)CB * CK * CNP * 2);  // [B,K,NP]
  _Float16* w16    = (_Float16*)alloc((size_t)CB * CTQ * CNP * 2); // [B,TQ,NP]
  _Float16* wt16   = (_Float16*)alloc((size_t)CB * CNP * CTQ * 2); // [B,NP,TQ]
  float*    iscore = (float*)   alloc((size_t)CB * CNP * 4);
  float*    qscore = (float*)   alloc((size_t)CB * CTQ * 4);

  auto blocks = [](long n) { return (unsigned)((n + 255) / 256); };

  // 1) Convert operands to f16 (weights transposed so GEMM B is always B^T)
  long nq = (long)CB * CTQ * CE;
  cvt_f32_f16<<<blocks(nq), 256, 0, stream>>>(ques_feat, qf16, nq);
  long ni = (long)CB * CNP * CE;
  build_imgT<<<blocks(ni), 256, 0, stream>>>(img_feat, imgT16);
  cvt_transpose_f16<<<blocks((long)CE * CE), 256, 0, stream>>>(Wc, WcT16, CE, CE);
  cvt_transpose_f16<<<blocks((long)CE * CK), 256, 0, stream>>>(Wq, WqT16, CE, CK);
  cvt_transpose_f16<<<blocks((long)CE * CK), 256, 0, stream>>>(Wi, WiT16, CE, CK);

  // 2) img_corr = imgT @ Wc + bc -> corr16 [B,NP,E] (row-major == B^T for step 5)
  gemm_wmma<EP_BIAS_F16><<<dim3((CNP + 63) / 64, CE / 128, CB), 256, 0, stream>>>(
      imgT16, CE, (long)CNP * CE, WcT16, CE, 0L, CNP, CE, CE, bc,
      corr16, CE, (long)CNP * CE, nullptr, 0, 0L, nullptr, 0, 0L, 0, nullptr, nullptr, 0);

  // 3) ques_embed = Q @ Wq + bq -> qe32 [B,TQ,K] + qeT16 [B,K,TQ]
  gemm_wmma<EP_BIAS_F32_F16T><<<dim3(CTQ / 64, CK / 128, CB), 256, 0, stream>>>(
      qf16, CE, (long)CTQ * CE, WqT16, CE, 0L, CTQ, CK, CE, bq,
      nullptr, 0, 0L, qe32, CK, (long)CTQ * CK, qeT16, CTQ, (long)CK * CTQ, 0,
      nullptr, nullptr, 0);

  // 4) img_embed = imgT @ Wi + bi -> ie32 [B,NP,K] + ieT16 [B,K,NP]
  gemm_wmma<EP_BIAS_F32_F16T><<<dim3((CNP + 63) / 64, CK / 128, CB), 256, 0, stream>>>(
      imgT16, CE, (long)CNP * CE, WiT16, CE, 0L, CNP, CK, CE, bi,
      nullptr, 0, 0L, ie32, CK, (long)CNP * CK, ieT16, CNP, (long)CK * CNP, 0,
      nullptr, nullptr, 0);

  // 5) weight = tanh(Q @ corr^T), zero padded cols -> w16 [B,TQ,NP] + wt16 [B,NP,TQ]
  gemm_wmma<EP_TANH_WEIGHT><<<dim3(CTQ / 64, (CNP + 127) / 128, CB), 256, 0, stream>>>(
      qf16, CE, (long)CTQ * CE, corr16, CE, (long)CNP * CE, CTQ, CNP, CE, nullptr,
      w16, CNP, (long)CTQ * CNP, nullptr, 0, 0L, wt16, CTQ, (long)CNP * CTQ, CN,
      nullptr, nullptr, 0);

  // 6) zero score accumulators
  zero_f32<<<blocks((long)CB * CNP), 256, 0, stream>>>(iscore, (long)CB * CNP);
  zero_f32<<<blocks((long)CB * CTQ), 256, 0, stream>>>(qscore, (long)CB * CTQ);

  // 7) image branch: img_score[n] = sum_k tanh(W^T@qe + ie)[n,k] * wia[k]
  gemm_wmma<EP_TANH_DOT><<<dim3((CNP + 63) / 64, CK / 128, CB), 256, 0, stream>>>(
      wt16, CTQ, (long)CNP * CTQ, qeT16, CTQ, (long)CK * CTQ, CNP, CK, CTQ, nullptr,
      nullptr, 0, 0L, ie32, CK, (long)CNP * CK, nullptr, 0, 0L, 0, wia, iscore, CNP);

  // 8) question branch: ques_score[q] = sum_k tanh(W@ie + qe)[q,k] * wqa[k]
  gemm_wmma<EP_TANH_DOT><<<dim3(CTQ / 64, CK / 128, CB), 256, 0, stream>>>(
      w16, CNP, (long)CTQ * CNP, ieT16, CNP, (long)CK * CNP, CTQ, CK, CNP, nullptr,
      nullptr, 0, 0L, qe32, CK, (long)CTQ * CK, nullptr, 0, 0L, 0, wqa, qscore, CTQ);

  // 9) softmax + attended features
  ques_finalize<<<CB, 256, 0, stream>>>(qscore, bqa, mask, ques_feat, out);
  img_finalize<<<CB, 256, 0, stream>>>(iscore, bia, img_feat, out);
}
